// MultiExpertLoraLinear_71708773974287
// MI455X (gfx1250) — matrix operand order repaired and checked by reading, not compiled, not run on hardware
//
#include <hip/hip_runtime.h>

// ---------------------------------------------------------------------------
// MultiExpertLoraLinear fused pipeline for gfx1250 (MI455X), wave32 + WMMA.
//   K0: low  = x @ Acat^T                  [8192 x 128]   (bf16-split WMMA)
//   K1: plow = masked-mean_t(low) per b    [4 x 128]
//   K2: weights = softmax(gate(plow))      [4 x 8]        (single block)
//   K3: out = x@W^T + b + (w*low)@(2*Bcat) [8192 x 2048]  (bf16-split WMMA,
//                                            K = 2048 + 128 fused)
// LDS holds pre-split bf16 hi/lo tiles (converted once at staging);
// double-buffered, one barrier per K-chunk; all 8 B-fragments preloaded so
// ds_load latency pipelines under the 24-WMMA burst.
// ---------------------------------------------------------------------------

typedef __attribute__((ext_vector_type(16))) __bf16 v16bf;
typedef __attribute__((ext_vector_type(8)))  __bf16 v8bf;
typedef __attribute__((ext_vector_type(4)))  __bf16 v4bf;
typedef __attribute__((ext_vector_type(8)))  float  v8f;

#define BW 40             // bf16 tile row stride (elems): 80B rows, conflict-free b128 gathers
#define SCALING 2.0f      // lora_alpha / r = 32/16

// --------------------------- staging helpers -------------------------------

// Load this thread's share of a 128x32 fp32 tile from a per-thread base
// pointer (rows stride 2048; per-it offset is 32 rows = 65536 floats).
__device__ __forceinline__ void load_tile_p(float4 r[4], const float* p) {
#pragma unroll
  for (int it = 0; it < 4; ++it)
    r[it] = *(const float4*)(p + (long)it * 65536);
}

// Split fp32 -> bf16 hi + bf16 lo(residual) and store into LDS tiles.
__device__ __forceinline__ void store_tile_bf(__bf16* hi, __bf16* lo,
                                              const float4 r[4], int t) {
#pragma unroll
  for (int it = 0; it < 4; ++it) {
    int idx = it * 256 + t;
    int row = idx >> 3;
    int kq  = (idx & 7) << 2;
    float f[4] = {r[it].x, r[it].y, r[it].z, r[it].w};
    v4bf h, l;
#pragma unroll
    for (int c = 0; c < 4; ++c) {
      float v = f[c];
      __bf16 hb = (__bf16)v;
      h[c] = hb;
      l[c] = (__bf16)(v - (float)hb);
    }
    *(v4bf*)(hi + row * BW + kq) = h;
    *(v4bf*)(lo + row * BW + kq) = l;
  }
}

// --------------------------- fragment gathers ------------------------------

// WMMA A-frag (16x32 bf16): lane 0-15 row=lane K={0..7,16..23};
//                           lane16-31 row=lane-16 K={8..15,24..31}.
__device__ __forceinline__ v16bf gather_a_bf(const __bf16* tile, int row0, int lane) {
  int m  = lane & 15;
  int k0 = (lane & 16) ? 8 : 0;
  const __bf16* p = tile + (row0 + m) * BW + k0;
  v8bf a0 = *(const v8bf*)(p);
  v8bf a1 = *(const v8bf*)(p + 16);
  return __builtin_shufflevector(a0, a1, 0, 1, 2, 3, 4, 5, 6, 7,
                                 8, 9, 10, 11, 12, 13, 14, 15);
}

// WMMA B-frag (32x16 bf16): lane 0-15 col=lane K={0..7,16..23}; lane16-31 K={8..15,24..31}.
__device__ __forceinline__ v16bf gather_b_bf(const __bf16* tile, int col0, int lane) {
  int n  = lane & 15;
  int ks = (lane & 16) ? 8 : 0;
  const __bf16* p = tile + (col0 + n) * BW + ks;
  v8bf b0 = *(const v8bf*)(p);
  v8bf b1 = *(const v8bf*)(p + 16);
  return __builtin_shufflevector(b0, b1, 0, 1, 2, 3, 4, 5, 6, 7,
                                 8, 9, 10, 11, 12, 13, 14, 15);
}

// One K=32 step for a wave's 32x64 output: 2x4 frags, 3 WMMAs each (split-bf16).
// All A and B fragments are gathered up front so the 24 ds_load_b128 pipeline
// under the 24 WMMAs instead of forcing s_wait_dscnt 0 per fragment.
__device__ __forceinline__ void mma_step_bf(const __bf16* xhi, const __bf16* xlo,
                                            const __bf16* whi, const __bf16* wlo,
                                            int lane, int waveRow, int waveCol,
                                            v8f acc[2][4]) {
  v16bf ahi[2], alo[2];
#pragma unroll
  for (int mf = 0; mf < 2; ++mf) {
    ahi[mf] = gather_a_bf(xhi, waveRow + mf * 16, lane);
    alo[mf] = gather_a_bf(xlo, waveRow + mf * 16, lane);
  }
  v16bf bhi[4], blo[4];
#pragma unroll
  for (int nf = 0; nf < 4; ++nf) {
    bhi[nf] = gather_b_bf(whi, waveCol + nf * 16, lane);
    blo[nf] = gather_b_bf(wlo, waveCol + nf * 16, lane);
  }
#pragma unroll
  for (int nf = 0; nf < 4; ++nf) {
#pragma unroll
    for (int mf = 0; mf < 2; ++mf) {
      acc[mf][nf] = __builtin_amdgcn_wmma_f32_16x16x32_bf16(
          false, ahi[mf], false, bhi[nf], (short)0, acc[mf][nf], false, false);
      acc[mf][nf] = __builtin_amdgcn_wmma_f32_16x16x32_bf16(
          false, alo[mf], false, bhi[nf], (short)0, acc[mf][nf], false, false);
      acc[mf][nf] = __builtin_amdgcn_wmma_f32_16x16x32_bf16(
          false, ahi[mf], false, blo[nf], (short)0, acc[mf][nf], false, false);
    }
  }
}

// ---------------------------------------------------------------------------
// K0: low[n, er] = sum_d x[n,d] * A_w[er, d]   (A_w viewed as [128, 2048])
// grid: 64 blocks (N/128), block: 256 threads (8 waves)
// ---------------------------------------------------------------------------
__global__ __launch_bounds__(256) void low_gemm_kernel(
    const float* __restrict__ x, const float* __restrict__ A_w,
    float* __restrict__ low) {
  // [buf][0=xhi,1=xlo,2=whi,3=wlo]
  __shared__ __bf16 sm[2][4][128 * BW];
  int t = threadIdx.x;
  int lane = t & 31, wave = t >> 5;
  int waveRow = (wave >> 1) * 32;
  int waveCol = (wave & 1) * 64;
  long rowBase = (long)blockIdx.x * 128;

  // per-thread running base pointers (advance 32 floats per K-chunk)
  long toff = (long)(t >> 3) * 2048 + ((t & 7) << 2);
  const float* px = x + rowBase * 2048 + toff;
  const float* pa = A_w + toff;

  v8f acc[2][4] = {};
  float4 rx[4], rw[4];
  load_tile_p(rx, px);
  load_tile_p(rw, pa);

  for (int kc = 0; kc < 64; ++kc) {
    int buf = kc & 1;
    store_tile_bf(sm[buf][0], sm[buf][1], rx, t);
    store_tile_bf(sm[buf][2], sm[buf][3], rw, t);
    __syncthreads();
    if (kc + 1 < 64) {
      px += 32; pa += 32;
      load_tile_p(rx, px);
      load_tile_p(rw, pa);
    }
    mma_step_bf(sm[buf][0], sm[buf][1], sm[buf][2], sm[buf][3],
                lane, waveRow, waveCol, acc);
  }

#pragma unroll
  for (int mf = 0; mf < 2; ++mf)
#pragma unroll
    for (int nf = 0; nf < 4; ++nf) {
      int col = waveCol + nf * 16 + (lane & 15);
      long r0 = rowBase + waveRow + mf * 16 + ((lane >> 4) << 3);
#pragma unroll
      for (int i = 0; i < 8; ++i)
        low[(r0 + i) * 128 + col] = acc[mf][nf][i];
    }
}

// ---------------------------------------------------------------------------
// K1: plow[b, er] = sum_t mask[b,t]*low[b*T+t, er] / max(sum_t mask, 1e-6)
// grid: 4 blocks, block: 128 threads
// ---------------------------------------------------------------------------
__global__ void pooled_kernel(const float* __restrict__ low,
                              const int* __restrict__ mask,
                              float* __restrict__ plow) {
  int b = blockIdx.x;
  int er = threadIdx.x;
  float acc = 0.f, ms = 0.f;
  for (int tt = 0; tt < 2048; ++tt) {
    float mv = (float)mask[b * 2048 + tt];
    acc += mv * low[((long)(b * 2048 + tt)) * 128 + er];
    ms += mv;
  }
  plow[b * 128 + er] = acc / fmaxf(ms, 1e-6f);
}

// ---------------------------------------------------------------------------
// K2: G[j,e,r] = sum_o gate_W[j, e*O+o] * B_w[e,o,r]; logits = plow . G;
//     weights = softmax(logits / tau).  Single block, 256 threads.
// ---------------------------------------------------------------------------
__global__ void gate_kernel(const float* __restrict__ plow_g,
                            const float* __restrict__ B_w,
                            const float* __restrict__ gate_W,
                            const float* __restrict__ gate_b,
                            const float* __restrict__ tau,
                            float* __restrict__ weights) {
  __shared__ float G[1024];     // [j=8][e=8][r=16]
  __shared__ float plow[512];   // [b=4][er=128]
  __shared__ float logits[32];  // [b=4][j=8]
  int t = threadIdx.x;

  for (int i = t; i < 512; i += 256) plow[i] = plow_g[i];
  for (int i = t; i < 1024; i += 256) {
    int j = i >> 7, e = (i >> 4) & 7, r = i & 15;
    const float* gw = gate_W + (long)j * 16384 + (long)e * 2048;
    const float* bw = B_w + (long)e * 32768 + r;
    float s = 0.f;
    for (int o = 0; o < 2048; ++o) s += gw[o] * bw[o * 16];
    G[i] = s;
  }
  __syncthreads();
  if (t < 32) {
    int b = t >> 3, j = t & 7;
    float s = 0.f;
    for (int k = 0; k < 128; ++k) s += plow[b * 128 + k] * G[j * 128 + k];
    logits[t] = (s + gate_b[j]) / tau[0];
  }
  __syncthreads();
  if (t < 4) {
    float mx = -1e30f;
    for (int j = 0; j < 8; ++j) mx = fmaxf(mx, logits[t * 8 + j]);
    float ex[8], s = 0.f;
    for (int j = 0; j < 8; ++j) { ex[j] = expf(logits[t * 8 + j] - mx); s += ex[j]; }
    for (int j = 0; j < 8; ++j) weights[t * 8 + j] = ex[j] / s;
  }
}

// ---------------------------------------------------------------------------
// K3: out[n,o] = sum_d x[n,d]*W[o,d] + b[o]
//             + sum_er (wts[b,e]*low[n,er]) * (SCALING*B_w[e,o,r])
// One GEMM with K = 2048 + 128. grid: (16 col tiles, 64 row tiles), block 256.
// ---------------------------------------------------------------------------
__global__ __launch_bounds__(256) void fused_out_kernel(
    const float* __restrict__ x, const float* __restrict__ base_W,
    const float* __restrict__ base_b, const float* __restrict__ low,
    const float* __restrict__ B_w, const float* __restrict__ wts,
    float* __restrict__ out) {
  __shared__ __bf16 sm[2][4][128 * BW];
  __shared__ float wlds[8];
  int t = threadIdx.x;
  int lane = t & 31, wave = t >> 5;
  int waveRow = (wave >> 1) * 32;
  int waveCol = (wave & 1) * 64;
  long rowBase = (long)blockIdx.y * 128;
  int colBase = blockIdx.x * 128;
  int bidx = (int)(rowBase >> 11);  // rowBase / T; tiles never straddle batches
  if (t < 8) wlds[t] = wts[bidx * 8 + t];

  // per-thread running base pointers for the main K region
  long toff = (long)(t >> 3) * 2048 + ((t & 7) << 2);
  const float* px = x + rowBase * 2048 + toff;
  const float* pw = base_W + (long)colBase * 2048 + toff;

  v8f acc[2][4] = {};
  float4 rx[4], rw[4];

  // fetch the LoRA region chunk kc2 (0..3) into registers
  auto fetch_lora = [&](int kc2) {
#pragma unroll
    for (int it = 0; it < 4; ++it) {
      int idx = it * 256 + t;
      int row = idx >> 3;
      int kq  = (idx & 7) << 2;
      int erB = kc2 * 32 + kq;  // multiple of 4 -> same expert within float4
      float wgt = wlds[erB >> 4];
      float4 v = *(const float4*)(low + (rowBase + row) * 128 + erB);
      v.x *= wgt; v.y *= wgt; v.z *= wgt; v.w *= wgt;
      rx[it] = v;
      int o = colBase + row;
      float4 u = *(const float4*)(B_w + (long)(erB >> 4) * 32768 +
                                  (long)o * 16 + (erB & 15));
      u.x *= SCALING; u.y *= SCALING; u.z *= SCALING; u.w *= SCALING;
      rw[it] = u;
    }
  };

  const int KT = 68;  // 64 base chunks + 4 LoRA chunks
  load_tile_p(rx, px);
  load_tile_p(rw, pw);
  for (int kc = 0; kc < KT; ++kc) {
    int buf = kc & 1;
    store_tile_bf(sm[buf][0], sm[buf][1], rx, t);
    store_tile_bf(sm[buf][2], sm[buf][3], rw, t);
    __syncthreads();
    if (kc + 1 < 64) {
      px += 32; pw += 32;
      load_tile_p(rx, px);
      load_tile_p(rw, pw);
    } else if (kc + 1 < KT) {
      fetch_lora(kc + 1 - 64);
    }
    mma_step_bf(sm[buf][0], sm[buf][1], sm[buf][2], sm[buf][3],
                lane, waveRow, waveCol, acc);
  }

  // Epilogue: +bias, store. C/D layout: VGPR i -> M = i + 8*(lane>=16), N = lane%16.
#pragma unroll
  for (int mf = 0; mf < 2; ++mf)
#pragma unroll
    for (int nf = 0; nf < 4; ++nf) {
      int col = colBase + waveCol + nf * 16 + (lane & 15);
      float bias = base_b[col];
      long r0 = rowBase + waveRow + mf * 16 + ((lane >> 4) << 3);
#pragma unroll
      for (int i = 0; i < 8; ++i)
        out[(r0 + i) * 2048 + col] = acc[mf][nf][i] + bias;
    }
}

// ---------------------------------------------------------------------------
extern "C" void kernel_launch(void* const* d_in, const int* in_sizes, int n_in,
                              void* d_out, int out_size, void* d_ws, size_t ws_size,
                              hipStream_t stream) {
  const float* x      = (const float*)d_in[0];
  const int*   attn   = (const int*)d_in[1];
  const float* base_W = (const float*)d_in[2];
  const float* base_b = (const float*)d_in[3];
  const float* A_w    = (const float*)d_in[4];
  const float* B_w    = (const float*)d_in[5];
  const float* gate_W = (const float*)d_in[6];
  const float* gate_b = (const float*)d_in[7];
  const float* tau    = (const float*)d_in[8];
  float* out = (float*)d_out;

  float* low  = (float*)d_ws;              // 8192*128 floats = 4 MB
  float* plow = low + (long)8192 * 128;    // 512 floats
  float* wts  = plow + 512;                // 32 floats

  low_gemm_kernel<<<dim3(64), dim3(256), 0, stream>>>(x, A_w, low);
  pooled_kernel<<<dim3(4), dim3(128), 0, stream>>>(low, attn, plow);
  gate_kernel<<<1, dim3(256), 0, stream>>>(plow, B_w, gate_W, gate_b, tau, wts);
  fused_out_kernel<<<dim3(16, 64), dim3(256), 0, stream>>>(x, base_W, base_b,
                                                           low, B_w, wts, out);
}
